// TrxMeanEncoder_73753178407534
// MI455X (gfx1250) — matrix operand order repaired and checked by reading, not compiled, not run on hardware
//
#include <hip/hip_runtime.h>

typedef __attribute__((ext_vector_type(16))) _Float16 v16h;
typedef __attribute__((ext_vector_type(8)))  _Float16 v8h;
typedef __attribute__((ext_vector_type(8)))  float    v8f;

#define Bn   16384
#define Tt   1024
#define VM   400
#define VT   100
#define KPM  416   // mcc K padded to 13*32 (u32 hist pitch)
#define KPT  128   // trx K padded to 4*32  (u32 hist pitch)
#define PHM  424   // f16 mirror pitch (16B-aligned rows, non-pow2 bank stride)
#define PHT  136
#define KTM  13
#define KTT  4
#define NTM  25    // 400/16 output col tiles (mcc)
#define NTT  7     // ceil(100/16) output col tiles (trx)
#define OUTC 501
#define WPACK_MCC_HALVES (KTM * NTM * 32 * 16)

// Pre-swizzle W (f32, row-major [V][V], row=k, col=n) into per-lane WMMA
// B-fragment layout, f16, zero-padded to Kp x (nT*16).
// Fragment (kt,nt): lane l<16 holds B[kt*32+0..15][nt*16+l],
//                   lane l>=16 holds B[kt*32+16..31][nt*16+(l-16)].
__global__ void pack_w_kernel(const float* __restrict__ W, _Float16* __restrict__ dst,
                              int V, int kT, int nT) {
  int tid = blockIdx.x * blockDim.x + threadIdx.x;
  int total = kT * nT * 32;
  if (tid >= total) return;
  int lane = tid & 31;
  int tile = tid >> 5;
  int nt = tile % nT;
  int kt = tile / nT;
  int n  = nt * 16 + (lane & 15);
  int k0 = kt * 32 + (lane >> 4) * 16;
  v16h v;
#pragma unroll
  for (int i = 0; i < 16; ++i) {
    int k = k0 + i;
    float w = (k < V && n < V) ? W[k * V + n] : 0.0f;
    v[i] = (_Float16)w;
  }
  *(((v16h*)dst) + (size_t)tile * 32 + lane) = v;
}

// One block per 16-row slab: LDS histograms + ragged log-mean + fused WMMA GEMM.
__global__ void __launch_bounds__(256)
trx_mean_encoder_fused(const int* __restrict__ mcc, const int* __restrict__ trx,
                       const float* __restrict__ amt, const int* __restrict__ slens,
                       const _Float16* __restrict__ wpM, const _Float16* __restrict__ wpT,
                       float* __restrict__ out) {
  __shared__ unsigned histM[16 * KPM];   // integer token counts (exact)
  __shared__ unsigned histT[16 * KPT];
  __shared__ _Float16 hm16[16 * PHM];    // f16 mirror for WMMA A-fragments
  __shared__ _Float16 ht16[16 * PHT];
  __shared__ float    means[16];

  const int tid  = threadIdx.x;
  const int lane = tid & 31;
  const int wave = tid >> 5;
  const int b0   = blockIdx.x * 16;

  for (int i = tid; i < 16 * KPM; i += 256) histM[i] = 0u;
  for (int i = tid; i < 16 * KPT; i += 256) histT[i] = 0u;
  __syncthreads();

  // ---- Phase 1: histogram + LogScaler ragged mean (each wave owns 2 rows) ----
  for (int rr = 0; rr < 2; ++rr) {
    const int row  = wave * 2 + rr;
    const int g    = b0 + row;
    const int sl   = slens[g];
    const int base = g * Tt;
    const uint4*  m4 = (const uint4*) (mcc + base);
    const uint4*  t4 = (const uint4*) (trx + base);
    const float4* a4 = (const float4*)(amt + base);
    unsigned* hM = &histM[row * KPM];
    unsigned* hT = &histT[row * KPT];
    float s = 0.0f;
#pragma unroll 2
    for (int i = 0; i < Tt / 128; ++i) {   // 4 tokens/lane/iter, b128 loads
      const int vi   = i * 32 + lane;
      const int tok0 = vi * 4;
      const uint4  mq = m4[vi];
      const uint4  tq = t4[vi];
      const float4 aq = a4[vi];
      atomicAdd(&hM[mq.x], 1u); atomicAdd(&hM[mq.y], 1u);
      atomicAdd(&hM[mq.z], 1u); atomicAdd(&hM[mq.w], 1u);
      atomicAdd(&hT[tq.x], 1u); atomicAdd(&hT[tq.y], 1u);
      atomicAdd(&hT[tq.z], 1u); atomicAdd(&hT[tq.w], 1u);
      const float v0 = copysignf(log1pf(fabsf(aq.x)), aq.x);
      const float v1 = copysignf(log1pf(fabsf(aq.y)), aq.y);
      const float v2 = copysignf(log1pf(fabsf(aq.z)), aq.z);
      const float v3 = copysignf(log1pf(fabsf(aq.w)), aq.w);
      if (tok0 + 0 < sl) s += v0;
      if (tok0 + 1 < sl) s += v1;
      if (tok0 + 2 < sl) s += v2;
      if (tok0 + 3 < sl) s += v3;
    }
#pragma unroll
    for (int off = 16; off > 0; off >>= 1) s += __shfl_down(s, off);
    if (lane == 0) means[row] = s / (float)sl;
  }
  __syncthreads();

  // ---- Phase 1b: one-time u32 -> f16 conversion of the count matrices ----
  for (int i = tid; i < 16 * PHM; i += 256) {
    const int r = i / PHM, k = i - r * PHM;
    hm16[i] = (_Float16)(float)(k < KPM ? histM[r * KPM + k] : 0u);
  }
  for (int i = tid; i < 16 * PHT; i += 256) {
    const int r = i / PHT, k = i - r * PHT;
    ht16[i] = (_Float16)(float)(k < KPT ? histT[r * KPT + k] : 0u);
  }
  __syncthreads();

  if (tid < 16) out[(size_t)(b0 + tid) * OUTC + (OUTC - 1)] = means[tid];

  // ---- Phase 2: (hist/T) @ W via v_wmma_f32_16x16x32_f16 ----
  // 32 n-tiles total (25 mcc + 7 trx); each wave computes 4 (wave-uniform ctrl).
  const float scale = 1.0f / (float)Tt;
  const int m     = lane & 15;
  const int khalf = (lane >> 4) * 16;    // A layout: lanes 16-31 hold K+16 half
  for (int j = 0; j < 4; ++j) {
    const int  tg  = wave + j * 8;
    const bool isM = (tg < NTM);
    const int nt          = isM ? tg   : tg - NTM;
    const _Float16* hst   = isM ? hm16 : ht16;
    const int pitch       = isM ? PHM  : PHT;
    const int kTiles      = isM ? KTM  : KTT;
    const int nT          = isM ? NTM  : NTT;
    const int colOff      = isM ? 0    : VM;
    const int nCols       = isM ? VM   : VT;
    const _Float16* wp    = isM ? wpM  : wpT;

    v8f c = {};
    for (int kt = 0; kt < kTiles; ++kt) {
      // A fragment: row m, K = kt*32 + {khalf..khalf+15}
      const _Float16* rp = hst + m * pitch + kt * 32 + khalf;
      const v8h lo = *(const v8h*)(rp);        // ds_load_b128
      const v8h hi = *(const v8h*)(rp + 8);    // ds_load_b128
      const v16h a = __builtin_shufflevector(lo, hi,
          0, 1, 2, 3, 4, 5, 6, 7, 8, 9, 10, 11, 12, 13, 14, 15);
      // B fragment: one contiguous 32B load from the packed buffer
      const v16h bf = *(((const v16h*)wp) + (size_t)(kt * nT + nt) * 32 + lane);
      // (neg_a, A, neg_b, B, c_mod, C, reuse_a, reuse_b)
      c = __builtin_amdgcn_wmma_f32_16x16x32_f16(false, a, false, bf,
                                                 (short)0, c, false, false);
    }
    // C/D layout: VGPR r -> row r (lanes 0-15) / row r+8 (lanes 16-31), col = lane&15
    const int col  = nt * 16 + (lane & 15);
    const int rAdd = (lane >> 4) * 8;
    if (col < nCols) {
#pragma unroll
      for (int r = 0; r < 8; ++r) {
        out[(size_t)(b0 + r + rAdd) * OUTC + colOff + col] = c[r] * scale;
      }
    }
  }
}

extern "C" void kernel_launch(void* const* d_in, const int* in_sizes, int n_in,
                              void* d_out, int out_size, void* d_ws, size_t ws_size,
                              hipStream_t stream) {
  const int*   mcc = (const int*)  d_in[0];
  const int*   trx = (const int*)  d_in[1];
  const float* amt = (const float*)d_in[2];
  const int*   sl  = (const int*)  d_in[3];
  const float* Wm  = (const float*)d_in[4];
  const float* Wt  = (const float*)d_in[5];
  float* out = (float*)d_out;

  _Float16* wpM = (_Float16*)d_ws;                                         // 332,800 B
  _Float16* wpT = (_Float16*)((char*)d_ws + (size_t)WPACK_MCC_HALVES * 2); // +28,672 B

  const int thrM = KTM * NTM * 32;  // 10400
  const int thrT = KTT * NTT * 32;  // 896
  pack_w_kernel<<<(thrM + 255) / 256, 256, 0, stream>>>(Wm, wpM, VM, KTM, NTM);
  pack_w_kernel<<<(thrT + 255) / 256, 256, 0, stream>>>(Wt, wpT, VT, KTT, NTT);

  trx_mean_encoder_fused<<<Bn / 16, 256, 0, stream>>>(mcc, trx, amt, sl, wpM, wpT, out);
}
